// GCNPredictor_23338852286551
// MI455X (gfx1250) — matrix operand order repaired and checked by reading, not compiled, not run on hardware
//
#include <hip/hip_runtime.h>
#include <hip/hip_bf16.h>

#define BZ 256
#define N 360
#define NH 4
#define HD 8
#define NP1 361
#define POOL_L 252
#define MROWS (BZ * N)      // 92160
#define KP 384              // padded K (360 -> 384)
#define NPAD 384            // padded N (360 -> 384, 6 groups of 4 tiles)

typedef _Float16 h8  __attribute__((ext_vector_type(8)));
typedef _Float16 h16 __attribute__((ext_vector_type(16)));
typedef float    f8v __attribute__((ext_vector_type(8)));

// ---------------- workspace arena (bytes) ----------------
constexpr size_t OFF_ROWSUM = 0;                       // 92160*4        (reused later for score)
constexpr size_t OFF_STATS  = 0x60000;                 // 1504 floats
constexpr size_t OFF_W0     = 0x70000;                 // 384*384*2
constexpr size_t OFF_W1     = 0xC0000;
constexpr size_t OFF_W2     = 0x110000;                // 64*384*2
constexpr size_t OFF_A      = 0x200000;                // f16 A: 92160*384*2 (~70.8MB) ; reused for H1
constexpr size_t OFF_C      = 0x4600000;               // f32 C: 92160*384*4 (~141.6MB); reused below
constexpr size_t OFF_X      = 0xD000000;               // f32 X: 92160*360*4 (~132.7MB)
constexpr size_t OFF_XIN    = OFF_C + 0x600000;        // 256*361*16*4
constexpr size_t OFF_Q      = OFF_C + 0x1000000;       // 256*361*32*4
constexpr size_t OFF_K      = OFF_C + 0x1C00000;
constexpr size_t OFF_V      = OFF_C + 0x2800000;
constexpr size_t OFF_OUT2   = OFF_C + 0x3400000;       // 256*361*16*4

// ---------------- utility kernels ----------------
__global__ void k_zero(float* p, int n) {
  int i = blockIdx.x * blockDim.x + threadIdx.x;
  if (i < n) p[i] = 0.f;
}

__global__ void k_rowsum(const float* __restrict__ m, float* __restrict__ rowsum) {
  int i = blockIdx.x * blockDim.x + threadIdx.x;
  if (i >= MROWS) return;
  const float* r = m + (size_t)i * N;
  float s = 0.f;
  for (int j = 0; j < N; ++j) s += r[j];
  rowsum[i] = s;
}

// Bt[n*KP + k] = W[k*Ncols + n]  (transposed, zero-padded)
__global__ void k_wcvt(const float* __restrict__ W, _Float16* __restrict__ Bt,
                       int Kdim, int Ncols, int Npadded) {
  int idx = blockIdx.x * blockDim.x + threadIdx.x;
  if (idx >= Npadded * KP) return;
  int n = idx / KP, k = idx - n * KP;
  float v = (k < Kdim && n < Ncols) ? W[(size_t)k * Ncols + n] : 0.f;
  Bt[idx] = (_Float16)v;
}

// Ah[i*KP + k] = f16(rowsum[i] * X[i*N + k]), zero-padded K
__global__ void k_buildA(const float* __restrict__ X, const float* __restrict__ rowsum,
                         _Float16* __restrict__ Ah) {
  int idx = blockIdx.x * blockDim.x + threadIdx.x;
  if (idx >= MROWS * KP) return;
  int i = idx / KP, k = idx - i * KP;
  float v = (k < N) ? rowsum[i] * X[(size_t)i * N + k] : 0.f;
  Ah[idx] = (_Float16)v;
}

// ---------------- WMMA GEMM: each wave computes a 16x64 strip of C (4 N-tiles) ----------------
// A fragment loaded once per K-chunk, reused by 4 WMMAs -> better wmma:load issue balance.
__global__ void k_gemm4(const _Float16* __restrict__ A, const _Float16* __restrict__ Bt,
                        float* __restrict__ C, int Mtiles, int Ngroups, int Np) {
  int wid  = blockIdx.x * (blockDim.x >> 5) + (threadIdx.x >> 5);
  int lane = threadIdx.x & 31;
  if (wid >= Mtiles * Ngroups) return;             // wave-uniform exit: EXEC stays all-ones
  int tm = wid / Ngroups, tg = wid - tm * Ngroups;
  int l15 = lane & 15, hi = lane >> 4;

  // A frag: lanes 0-15 row m, K = kk..kk+15; lanes 16-31: K = kk+16..kk+31
  const _Float16* pa = A  + (size_t)(tm * 16 + l15) * KP + (hi << 4);
  // B frags: lanes 0-15 col n, K = kk..kk+15; lanes 16-31: K = kk+16..kk+31 (Bt is N-major)
  const _Float16* pb = Bt + (size_t)(tg * 64 + l15) * KP + (hi << 4);
  const size_t bstep = (size_t)16 * KP;            // next N-tile of B

  f8v acc0 = {0.f,0.f,0.f,0.f,0.f,0.f,0.f,0.f};
  f8v acc1 = acc0, acc2 = acc0, acc3 = acc0;

#pragma unroll 2
  for (int kk = 0; kk < KP; kk += 32) {
    h8 alo = *(const h8*)(pa + kk);
    h8 ahi = *(const h8*)(pa + kk + 8);
    h16 a  = __builtin_shufflevector(alo, ahi, 0,1,2,3,4,5,6,7,8,9,10,11,12,13,14,15);
    h16 b0 = *(const h16*)(pb + kk);
    h16 b1 = *(const h16*)(pb + bstep + kk);
    h16 b2 = *(const h16*)(pb + 2 * bstep + kk);
    h16 b3 = *(const h16*)(pb + 3 * bstep + kk);
    if (kk + 32 < KP) {
      __builtin_prefetch(pa + kk + 32, 0, 0);
      __builtin_prefetch(pb + kk + 32, 0, 0);
    }
    acc0 = __builtin_amdgcn_wmma_f32_16x16x32_f16(false, a, false, b0, (short)0, acc0, false, false);
    acc1 = __builtin_amdgcn_wmma_f32_16x16x32_f16(false, a, false, b1, (short)0, acc1, false, false);
    acc2 = __builtin_amdgcn_wmma_f32_16x16x32_f16(false, a, false, b2, (short)0, acc2, false, false);
    acc3 = __builtin_amdgcn_wmma_f32_16x16x32_f16(false, a, false, b3, (short)0, acc3, false, false);
  }
  // C/D layout: lanes 0-15 -> N=lane, M = r; lanes 16-31 -> N=lane-16, M = 8+r
  int cm0 = tm * 16 + (hi << 3);
  int cn  = tg * 64 + l15;
#pragma unroll
  for (int r = 0; r < 8; ++r) {
    size_t rowoff = (size_t)(cm0 + r) * Np;
    C[rowoff + cn]      = acc0[r];
    C[rowoff + cn + 16] = acc1[r];
    C[rowoff + cn + 32] = acc2[r];
    C[rowoff + cn + 48] = acc3[r];
  }
}

// ---------------- fused bias + lrelu + residual + column stats ----------------
__global__ void k_post(float* __restrict__ C, int Np, int Ncols,
                       const float* __restrict__ bias, const float* __restrict__ res, int resStride,
                       float* __restrict__ sum, float* __restrict__ sumsq,
                       int rowsPerBlock, int doLrelu) {
  int j  = threadIdx.x;
  int r0 = blockIdx.x * rowsPerBlock;
  if (j >= Ncols) return;
  float s = 0.f, q = 0.f;
  for (int r = 0; r < rowsPerBlock; ++r) {
    size_t row = (size_t)(r0 + r);
    float y = C[row * Np + j];
    if (bias) y += bias[j];
    if (doLrelu) y = (y > 0.f) ? y : 0.2f * y;
    if (res) y += res[row * resStride + j];
    C[row * Np + j] = y;
    s += y; q += y * y;
  }
  atomicAdd(&sum[j], s);
  atomicAdd(&sumsq[j], q);
}

__global__ void k_bnapply(const float* __restrict__ C, int Np, int Ncols,
                          const float* __restrict__ sum, const float* __restrict__ sumsq,
                          const float* __restrict__ g, const float* __restrict__ b,
                          float* __restrict__ Xo, int outStride) {
  int idx = blockIdx.x * blockDim.x + threadIdx.x;
  if (idx >= MROWS * Ncols) return;
  int i = idx / Ncols, j = idx - i * Ncols;
  float inv = 1.f / (float)MROWS;
  float mu  = sum[j] * inv;
  float var = sumsq[j] * inv - mu * mu;
  float y = C[(size_t)i * Np + j];
  Xo[(size_t)i * outStride + j] = (y - mu) * rsqrtf(var + 1e-5f) * g[j] + b[j];
}

// ---------------- GCN2 branch ----------------
__global__ void k_gcn2mid(const float* __restrict__ Cp, const float* __restrict__ b1,
                          float* __restrict__ H1) {
  int idx = blockIdx.x * blockDim.x + threadIdx.x;
  if (idx >= MROWS * 64) return;
  int d = idx & 63;
  float y = Cp[idx] + b1[d];
  H1[idx] = (y > 0.f) ? y : 0.2f * y;
}

__global__ void k_gcn2out(const float* __restrict__ H1, const float* __restrict__ w2,
                          const float* __restrict__ b2, const float* __restrict__ X,
                          const float* __restrict__ scw, const float* __restrict__ scb,
                          float* __restrict__ X16) {
  int idx = blockIdx.x * blockDim.x + threadIdx.x;
  if (idx >= MROWS * 16) return;
  int d = idx & 15, i = idx >> 4;
  float a = b2[d];
  const float* hp = H1 + (size_t)i * 64;
  for (int c = 0; c < 64; ++c) a += hp[c] * w2[c * 16 + d];
  a = (a > 0.f) ? a : 0.2f * a;
  float sc = scb[d];
  const float* xp = X + (size_t)i * N;
  for (int j = 0; j < N; ++j) sc += xp[j] * scw[j * 16 + d];
  sc = (sc > 0.f) ? sc : 0.2f * sc;
  X16[idx] = a + sc;
}

// BN2 + prepend cls row -> xin[b, 0..360, 16]
__global__ void k_bn2apply(const float* __restrict__ X16, const float* __restrict__ sum,
                           const float* __restrict__ sumsq, const float* __restrict__ g,
                           const float* __restrict__ b, const float* __restrict__ cls,
                           float* __restrict__ xin) {
  int idx = blockIdx.x * blockDim.x + threadIdx.x;
  if (idx >= BZ * NP1 * 16) return;
  int j = idx & 15, row = idx >> 4;
  int bz = row / NP1, r = row - bz * NP1;
  float v;
  if (r == 0) v = cls[j];
  else {
    int i = bz * N + (r - 1);
    float inv = 1.f / (float)MROWS;
    float mu  = sum[j] * inv;
    float var = sumsq[j] * inv - mu * mu;
    v = (X16[(size_t)i * 16 + j] - mu) * rsqrtf(var + 1e-5f) * g[j] + b[j];
  }
  xin[idx] = v;
}

// ---------------- attention ----------------
__global__ void k_qkv(const float* __restrict__ xin,
                      const float* __restrict__ wq, const float* __restrict__ bq,
                      const float* __restrict__ wk, const float* __restrict__ bk,
                      const float* __restrict__ wv, const float* __restrict__ bv,
                      float* __restrict__ q, float* __restrict__ k, float* __restrict__ v) {
  int idx = blockIdx.x * blockDim.x + threadIdx.x;
  if (idx >= BZ * NP1 * 32) return;
  int d = idx & 31, row = idx >> 5;
  const float* xr = xin + (size_t)row * 16;
  float aq = bq[d], ak = bk[d], av = bv[d];
  for (int c = 0; c < 16; ++c) {
    float x = xr[c];
    aq += x * wq[c * 32 + d];
    ak += x * wk[c * 32 + d];
    av += x * wv[c * 32 + d];
  }
  q[idx] = aq; k[idx] = ak; v[idx] = av;
}

__global__ void k_attn(const float* __restrict__ q, const float* __restrict__ k,
                       float* __restrict__ attn) {
  __shared__ float red[512];
  int bid = blockIdx.x;                      // (b*NH + h)*NP1 + qr
  int qr = bid % NP1, bh = bid / NP1;
  int h = bh % NH, b = bh / NH;
  int t = threadIdx.x;
  const float* qp = q + (size_t)(b * NP1 + qr) * 32 + h * HD;
  float logit = -3.0e38f;
  if (t < NP1) {
    const float* kp = k + (size_t)(b * NP1 + t) * 32 + h * HD;
    float s = 0.f;
#pragma unroll
    for (int d = 0; d < HD; ++d) s += qp[d] * kp[d];
    logit = s * 0.35355339059327373f;        // 1/sqrt(8)
  }
  red[t] = logit; __syncthreads();
  for (int st = 256; st > 0; st >>= 1) {
    if (t < st) { float o = red[t + st]; if (o > red[t]) red[t] = o; }
    __syncthreads();
  }
  float mx = red[0]; __syncthreads();
  float e = (t < NP1) ? __expf(logit - mx) : 0.f;
  red[t] = e; __syncthreads();
  for (int st = 256; st > 0; st >>= 1) {
    if (t < st) red[t] += red[t + st];
    __syncthreads();
  }
  float inv = 1.f / red[0];
  if (t < NP1) attn[(size_t)bid * NP1 + t] = e * inv;
}

__global__ void k_score(const float* __restrict__ attn, float* __restrict__ score,
                        float* __restrict__ sig_out) {
  int idx = blockIdx.x * blockDim.x + threadIdx.x;
  if (idx >= BZ * N) return;
  int b = idx / N, t = idx - b * N + 1;
  float s = 0.f;
  for (int h = 0; h < NH; ++h) s += attn[((size_t)(b * NH + h) * NP1) * NP1 + t];
  score[idx] = s;
  sig_out[idx] = 1.f / (1.f + __expf(-s));
}

__global__ void k_av(const float* __restrict__ attn, const float* __restrict__ v,
                     const float* __restrict__ wo, const float* __restrict__ bo,
                     float* __restrict__ out16) {
  __shared__ float ov[32];
  int bid = blockIdx.x;                      // b*NP1 + qr
  int qr = bid % NP1, b = bid / NP1;
  int t = threadIdx.x;
  if (t < 32) {
    int h = t >> 3;
    const float* ap = attn + ((size_t)(b * NH + h) * NP1 + qr) * NP1;
    const float* vp = v + (size_t)b * NP1 * 32 + t;
    float s = 0.f;
    for (int kx = 0; kx < NP1; ++kx) s += ap[kx] * vp[(size_t)kx * 32];
    ov[t] = s;
  }
  __syncthreads();
  if (t < 16) {
    float s = bo[t];
    for (int c = 0; c < 32; ++c) s += ov[c] * wo[c * 16 + t];
    out16[(size_t)bid * 16 + t] = s;
  }
}

// ---------------- top-k pooling (order-invariant) + MLP head ----------------
__global__ void k_pool(const float* __restrict__ score, const float* __restrict__ xb,
                       const float* __restrict__ f1w, const float* __restrict__ f1b,
                       const float* __restrict__ f2w, const float* __restrict__ f2b,
                       const float* __restrict__ f3w, const float* __restrict__ f3b,
                       float* __restrict__ logits) {
  __shared__ float s[384];
  __shared__ float red[512];
  __shared__ float pooled[16];
  __shared__ float h1[64], h2[32];
  int b = blockIdx.x, t = threadIdx.x;
  if (t < N) s[t] = score[b * N + t];
  else if (t < 384) s[t] = -3.0e38f;
  if (t < 16) pooled[t] = 0.f;
  __syncthreads();
  int sel = 0; float my = -3.0e38f;
  if (t < N) {
    my = s[t];
    int cnt = 0;
    for (int j = 0; j < N; ++j) {
      float o = s[j];
      cnt += (o > my) || (o == my && j < t);
    }
    sel = (cnt < POOL_L);
  }
  red[t] = (t < N) ? my : -3.0e38f; __syncthreads();
  for (int st = 256; st > 0; st >>= 1) {
    if (t < st) { float o = red[t + st]; if (o > red[t]) red[t] = o; }
    __syncthreads();
  }
  float mx = red[0]; __syncthreads();
  float e = sel ? __expf(my - mx) : 0.f;
  red[t] = e; __syncthreads();
  for (int st = 256; st > 0; st >>= 1) {
    if (t < st) red[t] += red[t + st];
    __syncthreads();
  }
  float w = e / red[0];
  __syncthreads();
  if (sel && w != 0.f) {
    const float* xp = xb + ((size_t)(b * NP1) + 1 + t) * 16;
    for (int d = 0; d < 16; ++d) atomicAdd(&pooled[d], w * xp[d]);
  }
  __syncthreads();
  if (t < 64) {
    float a = f1b[t];
    for (int c = 0; c < 16; ++c) a += pooled[c] * f1w[c * 64 + t];
    h1[t] = (a > 0.f) ? a : 0.2f * a;
  }
  __syncthreads();
  if (t < 32) {
    float a = f2b[t];
    for (int c = 0; c < 64; ++c) a += h1[c] * f2w[c * 32 + t];
    h2[t] = (a > 0.f) ? a : 0.2f * a;
  }
  __syncthreads();
  if (t < 2) {
    float a = f3b[t];
    for (int c = 0; c < 32; ++c) a += h2[c] * f3w[c * 2 + t];
    logits[b * 2 + t] = a;
  }
}

// ---------------- host side ----------------
extern "C" void kernel_launch(void* const* d_in, const int* in_sizes, int n_in,
                              void* d_out, int out_size, void* d_ws, size_t ws_size,
                              hipStream_t stream) {
  (void)in_sizes; (void)n_in; (void)out_size; (void)ws_size;
  const float* m    = (const float*)d_in[0];
  const float* nf   = (const float*)d_in[1];
  const float* cls  = (const float*)d_in[2];
  const float* g0w  = (const float*)d_in[3];
  const float* g0b  = (const float*)d_in[4];
  const float* g1w  = (const float*)d_in[5];
  const float* g1b  = (const float*)d_in[6];
  const float* g2w1 = (const float*)d_in[7];
  const float* g2b1 = (const float*)d_in[8];
  const float* g2w2 = (const float*)d_in[9];
  const float* g2b2 = (const float*)d_in[10];
  const float* scw  = (const float*)d_in[11];
  const float* scb  = (const float*)d_in[12];
  const float* bn0g = (const float*)d_in[13];
  const float* bn0b = (const float*)d_in[14];
  const float* bn1g = (const float*)d_in[15];
  const float* bn1b = (const float*)d_in[16];
  const float* bn2g = (const float*)d_in[17];
  const float* bn2b = (const float*)d_in[18];
  const float* wq   = (const float*)d_in[19];
  const float* bq   = (const float*)d_in[20];
  const float* wk   = (const float*)d_in[21];
  const float* bk   = (const float*)d_in[22];
  const float* wv   = (const float*)d_in[23];
  const float* bv   = (const float*)d_in[24];
  const float* wo   = (const float*)d_in[25];
  const float* bo   = (const float*)d_in[26];
  const float* f1w  = (const float*)d_in[27];
  const float* f1b  = (const float*)d_in[28];
  const float* f2w  = (const float*)d_in[29];
  const float* f2b  = (const float*)d_in[30];
  const float* f3w  = (const float*)d_in[31];
  const float* f3b  = (const float*)d_in[32];

  float* out = (float*)d_out;
  float* out_logits = out;
  float* out_sig    = out + 512;
  float* out_attn   = out + 512 + MROWS;

  char* ws = (char*)d_ws;
  float*    rowsum = (float*)(ws + OFF_ROWSUM);
  float*    stats  = (float*)(ws + OFF_STATS);     // [sum0|sq0|sum1|sq1|sum2|sq2]
  _Float16* W0t    = (_Float16*)(ws + OFF_W0);
  _Float16* W1t    = (_Float16*)(ws + OFF_W1);
  _Float16* W2t    = (_Float16*)(ws + OFF_W2);
  _Float16* Ah     = (_Float16*)(ws + OFF_A);
  float*    Cp     = (float*)(ws + OFF_C);
  float*    X      = (float*)(ws + OFF_X);
  float*    H1     = (float*)(ws + OFF_A);         // reuse (A dead after last GEMM)
  float*    X16    = (float*)(ws + OFF_C);         // reuse (C dead after gcn2mid)
  float*    XIN    = (float*)(ws + OFF_XIN);
  float*    Q      = (float*)(ws + OFF_Q);
  float*    K      = (float*)(ws + OFF_K);
  float*    V      = (float*)(ws + OFF_V);
  float*    OUT16  = (float*)(ws + OFF_OUT2);
  float*    SCORE  = (float*)(ws + OFF_ROWSUM);    // reuse (rowsum dead after last buildA)

  const int Mtiles = MROWS / 16;                   // 5760
  // prep
  k_zero<<<(1504 + 255) / 256, 256, 0, stream>>>(stats, 1504);
  k_rowsum<<<(MROWS + 255) / 256, 256, 0, stream>>>(m, rowsum);
  k_wcvt<<<(NPAD * KP + 255) / 256, 256, 0, stream>>>(g0w, W0t, N, N, NPAD);
  k_wcvt<<<(NPAD * KP + 255) / 256, 256, 0, stream>>>(g1w, W1t, N, N, NPAD);
  k_wcvt<<<(64 * KP + 255) / 256, 256, 0, stream>>>(g2w1, W2t, N, 64, 64);

  // GCN layer 0 (residual = 0); NPAD=384 -> 6 groups of 4 N-tiles
  k_buildA<<<(MROWS * KP + 255) / 256, 256, 0, stream>>>(nf, rowsum, Ah);
  k_gemm4<<<(Mtiles * 6) / 8, 256, 0, stream>>>(Ah, W0t, Cp, Mtiles, 6, NPAD);
  k_post<<<360, 384, 0, stream>>>(Cp, NPAD, N, g0b, nullptr, 0, stats, stats + 368, 256, 1);
  k_bnapply<<<(MROWS * N + 255) / 256, 256, 0, stream>>>(Cp, NPAD, N, stats, stats + 368, bn0g, bn0b, X, N);

  // GCN layer 1 (residual = X)
  k_buildA<<<(MROWS * KP + 255) / 256, 256, 0, stream>>>(X, rowsum, Ah);
  k_gemm4<<<(Mtiles * 6) / 8, 256, 0, stream>>>(Ah, W1t, Cp, Mtiles, 6, NPAD);
  k_post<<<360, 384, 0, stream>>>(Cp, NPAD, N, g1b, X, N, stats + 736, stats + 1104, 256, 1);
  k_bnapply<<<(MROWS * N + 255) / 256, 256, 0, stream>>>(Cp, NPAD, N, stats + 736, stats + 1104, bn1g, bn1b, X, N);

  // GCN layer 2 + shortcut + BN2 (Np=64 -> exactly 1 group of 4 tiles)
  k_buildA<<<(MROWS * KP + 255) / 256, 256, 0, stream>>>(X, rowsum, Ah);
  k_gemm4<<<(Mtiles * 1) / 8, 256, 0, stream>>>(Ah, W2t, Cp, Mtiles, 1, 64);
  k_gcn2mid<<<(MROWS * 64 + 255) / 256, 256, 0, stream>>>(Cp, g2b1, H1);
  k_gcn2out<<<(MROWS * 16 + 255) / 256, 256, 0, stream>>>(H1, g2w2, g2b2, X, scw, scb, X16);
  k_post<<<360, 64, 0, stream>>>(X16, 16, 16, nullptr, nullptr, 0, stats + 1472, stats + 1488, 256, 0);
  k_bn2apply<<<(BZ * NP1 * 16 + 255) / 256, 256, 0, stream>>>(X16, stats + 1472, stats + 1488, bn2g, bn2b, cls, XIN);

  // attention head
  k_qkv<<<(BZ * NP1 * 32 + 255) / 256, 256, 0, stream>>>(XIN, wq, bq, wk, bk, wv, bv, Q, K, V);
  k_attn<<<BZ * NH * NP1, 512, 0, stream>>>(Q, K, out_attn);
  k_score<<<(BZ * N + 255) / 256, 256, 0, stream>>>(out_attn, SCORE, out_sig);
  k_av<<<BZ * NP1, 64, 0, stream>>>(out_attn, V, wo, bo, OUT16);
  k_pool<<<BZ, 512, 0, stream>>>(SCORE, OUT16, f1w, f1b, f2w, f2b, f3w, f3b, out_logits);
}
